// AGNNNet_49993419325967
// MI455X (gfx1250) — compile-verified, matrix operand order checked by reading, not run on hardware
//
#include <hip/hip_runtime.h>
#include <hip/hip_bf16.h>

typedef __attribute__((ext_vector_type(16))) __bf16 v16bf;
typedef __attribute__((ext_vector_type(8)))  __bf16 v8bf;
typedef __attribute__((ext_vector_type(8)))  float  v8f;

#define F_IN 512
#define HID  16
#define NCLS 64

// ---- order-preserving float<->uint mapping for atomic max over floats ----
__device__ __forceinline__ unsigned fmap(float f) {
    unsigned u = __float_as_uint(f);
    return (u & 0x80000000u) ? ~u : (u | 0x80000000u);
}
__device__ __forceinline__ float funmap(unsigned u) {
    unsigned v = (u & 0x80000000u) ? (u & 0x7fffffffu) : ~u;
    return __uint_as_float(v);
}

// =====================================================================
// Kernel 1: h = relu(x @ W1 + b1), [N,512]x[512,16] via WMMA bf16 (f32 acc)
// One wave32 per 16x16 output tile; K swept in 32-steps (16 WMMA/tile).
// W1 staged in LDS transposed ([n][k], bf16) for contiguous B-fragment reads.
// =====================================================================
__global__ void k_gemm1(const float* __restrict__ x, const float* __restrict__ W1,
                        const float* __restrict__ b1, float* __restrict__ h, int nN) {
    __shared__ __bf16 Wl[HID * F_IN];   // 16 KB, layout [n*512 + k]
    int tid = threadIdx.x;
    for (int idx = tid; idx < F_IN * HID; idx += blockDim.x) {
        int k = idx >> 4, n = idx & 15;               // W1 is [k][n] row-major
        Wl[n * F_IN + k] = (__bf16)W1[idx];
    }
    __syncthreads();

    int wave = tid >> 5, lane = tid & 31;
    int tile = blockIdx.x * 8 + wave;
    int nTiles = nN >> 4;                              // nN is a multiple of 16
    if (tile >= nTiles) return;

    int base  = tile << 4;
    int lrow  = lane & 15;                             // A row (M) and B column (N)
    int khalf = (lane >> 4) << 3;                      // 0 or 8 (K sub-half per lane group)
    const float* xr = x + (size_t)(base + lrow) * F_IN;

    v8f acc = {};
    for (int kk = 0; kk < F_IN; kk += 32) {
        // A fragment: K = kk+khalf+0..7  and  kk+16+khalf+0..7 (ISA 16-bit A 16x32 layout)
        float fa[16];
        const float4* p0 = (const float4*)(xr + kk + khalf);
        float4 t0 = p0[0], t1 = p0[1];
        const float4* p1 = (const float4*)(xr + kk + 16 + khalf);
        float4 t2 = p1[0], t3 = p1[1];
        fa[0]=t0.x;  fa[1]=t0.y;  fa[2]=t0.z;  fa[3]=t0.w;
        fa[4]=t1.x;  fa[5]=t1.y;  fa[6]=t1.z;  fa[7]=t1.w;
        fa[8]=t2.x;  fa[9]=t2.y;  fa[10]=t2.z; fa[11]=t2.w;
        fa[12]=t3.x; fa[13]=t3.y; fa[14]=t3.z; fa[15]=t3.w;
        v16bf a, b;
        #pragma unroll
        for (int i = 0; i < 16; ++i) a[i] = (__bf16)fa[i];

        // B fragment: column n=lrow, same K pattern, contiguous in LDS
        v8bf b0 = *(const v8bf*)&Wl[lrow * F_IN + kk + khalf];
        v8bf b1v = *(const v8bf*)&Wl[lrow * F_IN + kk + 16 + khalf];
        #pragma unroll
        for (int i = 0; i < 8; ++i) { b[i] = b0[i]; b[8 + i] = b1v[i]; }

        acc = __builtin_amdgcn_wmma_f32_16x16x32_bf16(
            false, a, false, b, (short)0, acc, false, false);
    }

    // D layout: VGPR j -> M = j + 8*(lane>=16), N = lane&15
    float bn = b1[lrow];
    int mtop = (lane >> 4) << 3;
    #pragma unroll
    for (int j = 0; j < 8; ++j) {
        float v = fmaxf(acc[j] + bn, 0.0f);
        h[(size_t)(base + j + mtop) * HID + lrow] = v;
    }
}

// ===================== per-node L2 normalize (eps=1e-12) ====================
__global__ void k_norm(const float* __restrict__ hin, float* __restrict__ xn, int nN) {
    int i = blockIdx.x * blockDim.x + threadIdx.x;
    if (i >= nN) return;
    const float4* p = (const float4*)(hin + (size_t)i * HID);
    float4 v0 = p[0], v1 = p[1], v2 = p[2], v3 = p[3];
    float s = v0.x*v0.x + v0.y*v0.y + v0.z*v0.z + v0.w*v0.w
            + v1.x*v1.x + v1.y*v1.y + v1.z*v1.z + v1.w*v1.w
            + v2.x*v2.x + v2.y*v2.y + v2.z*v2.z + v2.w*v2.w
            + v3.x*v3.x + v3.y*v3.y + v3.z*v3.z + v3.w*v3.w;
    float r = 1.0f / fmaxf(sqrtf(s), 1e-12f);
    float4* q = (float4*)(xn + (size_t)i * HID);
    v0.x*=r; v0.y*=r; v0.z*=r; v0.w*=r;  q[0]=v0;
    v1.x*=r; v1.y*=r; v1.z*=r; v1.w*=r;  q[1]=v1;
    v2.x*=r; v2.y*=r; v2.z*=r; v2.w*=r;  q[2]=v2;
    v3.x*=r; v3.y*=r; v3.z*=r; v3.w*=r;  q[3]=v3;
}

// ============ per-layer init: m=-inf(mapped), denom=0, out=0 ================
__global__ void k_init(unsigned* __restrict__ m, float* __restrict__ denom,
                       float* __restrict__ out, int nN) {
    int i = blockIdx.x * blockDim.x + threadIdx.x;
    if (i >= nN) return;
    m[i] = 0x007FFFFFu;            // fmap(-inf)
    denom[i] = 0.0f;
    float4 z = {0.f, 0.f, 0.f, 0.f};
    float4* q = (float4*)(out + (size_t)i * HID);
    q[0] = z; q[1] = z; q[2] = z; q[3] = z;
}

// ===== Pass A: logit = beta*cos(xn[s],xn[d]); segment max via atomicMax =====
__global__ void k_edge_logit(const float* __restrict__ xn, const long long* __restrict__ ei,
                             const float* __restrict__ beta, unsigned* __restrict__ m,
                             float* __restrict__ elog, int nE, int nN) {
    int e = blockIdx.x * blockDim.x + threadIdx.x;
    if (e >= nE + nN) return;
    __builtin_prefetch(ei + e + 4096, 0, 1);   // global_prefetch_b8 (speculative OK)
    int s, d;
    if (e < nE) { s = (int)ei[e]; d = (int)ei[(size_t)nE + e]; }
    else        { s = d = e - nE; }            // self-loop
    const float4* a = (const float4*)(xn + (size_t)s * HID);
    const float4* b = (const float4*)(xn + (size_t)d * HID);
    float dot = 0.0f;
    #pragma unroll
    for (int i = 0; i < 4; ++i) {
        float4 u = a[i], v = b[i];
        dot += u.x*v.x + u.y*v.y + u.z*v.z + u.w*v.w;
    }
    float lg = beta[0] * dot;
    elog[e] = lg;
    atomicMax(&m[d], fmap(lg));
}

// ========== Pass B: e = exp(logit - m[d]); segment sum of e =================
__global__ void k_edge_exp(const long long* __restrict__ ei, const unsigned* __restrict__ m,
                           float* __restrict__ denom, float* __restrict__ elog,
                           int nE, int nN) {
    int e = blockIdx.x * blockDim.x + threadIdx.x;
    if (e >= nE + nN) return;
    int d = (e < nE) ? (int)ei[(size_t)nE + e] : (e - nE);
    float ee = __expf(elog[e] - funmap(m[d]));
    elog[e] = ee;
    atomicAdd(&denom[d], ee);
}

// ===== Pass C: alpha = e/denom[d]; out[d] += alpha * hin[s] (scatter) =======
__global__ void k_edge_agg(const long long* __restrict__ ei, const float* __restrict__ hin,
                           const float* __restrict__ denom, const float* __restrict__ elog,
                           float* __restrict__ out, int nE, int nN) {
    int e = blockIdx.x * blockDim.x + threadIdx.x;
    if (e >= nE + nN) return;
    int s, d;
    if (e < nE) { s = (int)ei[e]; d = (int)ei[(size_t)nE + e]; }
    else        { s = d = e - nE; }
    float alpha = elog[e] / denom[d];
    const float4* hs = (const float4*)(hin + (size_t)s * HID);
    float* ob = out + (size_t)d * HID;
    #pragma unroll
    for (int i = 0; i < 4; ++i) {
        float4 v = hs[i];
        atomicAdd(ob + i * 4 + 0, alpha * v.x);
        atomicAdd(ob + i * 4 + 1, alpha * v.y);
        atomicAdd(ob + i * 4 + 2, alpha * v.z);
        atomicAdd(ob + i * 4 + 3, alpha * v.w);
    }
}

// === out = log_softmax(h @ W2 + b2): one wave32 per node, 2 classes/lane ===
__global__ void k_out(const float* __restrict__ h, const float* __restrict__ W2,
                      const float* __restrict__ b2, float* __restrict__ out, int nN) {
    __shared__ float W2s[HID * NCLS];   // 4 KB
    __shared__ float b2s[NCLS];
    int tid = threadIdx.x;
    for (int idx = tid; idx < HID * NCLS; idx += blockDim.x) W2s[idx] = W2[idx];
    if (tid < NCLS) b2s[tid] = b2[tid];
    __syncthreads();

    int wave = tid >> 5, lane = tid & 31;
    int node = blockIdx.x * 8 + wave;
    if (node >= nN) return;

    float hv[HID];
    const float4* hp = (const float4*)(h + (size_t)node * HID);
    #pragma unroll
    for (int i = 0; i < 4; ++i) {
        float4 v = hp[i];
        hv[i*4+0]=v.x; hv[i*4+1]=v.y; hv[i*4+2]=v.z; hv[i*4+3]=v.w;
    }
    float acc0 = b2s[lane], acc1 = b2s[lane + 32];
    #pragma unroll
    for (int k = 0; k < HID; ++k) {
        acc0 += hv[k] * W2s[k * NCLS + lane];
        acc1 += hv[k] * W2s[k * NCLS + lane + 32];
    }
    // wave32 reductions
    float mx = fmaxf(acc0, acc1);
    #pragma unroll
    for (int off = 16; off > 0; off >>= 1) mx = fmaxf(mx, __shfl_xor(mx, off));
    float sm = __expf(acc0 - mx) + __expf(acc1 - mx);
    #pragma unroll
    for (int off = 16; off > 0; off >>= 1) sm += __shfl_xor(sm, off);
    float lse = mx + __logf(sm);
    out[(size_t)node * NCLS + lane]      = acc0 - lse;
    out[(size_t)node * NCLS + lane + 32] = acc1 - lse;
}

extern "C" void kernel_launch(void* const* d_in, const int* in_sizes, int n_in,
                              void* d_out, int out_size, void* d_ws, size_t ws_size,
                              hipStream_t stream) {
    const float*     x     = (const float*)d_in[0];
    const long long* ei    = (const long long*)d_in[1];   // int64 [2, E]
    const float*     W1    = (const float*)d_in[2];
    const float*     b1    = (const float*)d_in[3];
    const float*     W2    = (const float*)d_in[4];
    const float*     b2    = (const float*)d_in[5];
    const float*     beta1 = (const float*)d_in[6];
    const float*     beta2 = (const float*)d_in[7];

    int nN  = in_sizes[0] / F_IN;      // 100000 (multiple of 16)
    int nE  = in_sizes[1] / 2;         // 3,200,000
    int tot = nE + nN;                 // edges + self-loops

    // workspace carve-up (~33 MB)
    float*    ws    = (float*)d_ws;
    float*    P0    = ws;                               // layer input h
    float*    P1    = ws + (size_t)nN * HID;            // normalized features
    float*    P2    = ws + 2 * (size_t)nN * HID;        // aggregate output
    unsigned* mBuf  = (unsigned*)(ws + 3 * (size_t)nN * HID);
    float*    denom = (float*)(mBuf + nN);
    float*    elog  = denom + nN;                       // per-edge scratch

    dim3 blk(256);
    int tiles   = nN >> 4;
    int gGemm   = (tiles + 7) / 8;
    int gNode   = (nN + 255) / 256;
    int gEdge   = (tot + 255) / 256;
    int gOut    = (nN + 7) / 8;

    // h = relu(x @ W1 + b1)  -> P0
    k_gemm1<<<gGemm, blk, 0, stream>>>(x, W1, b1, P0, nN);

    // ---- AGNN layer 1: P0 -> P2 ----
    k_norm      <<<gNode, blk, 0, stream>>>(P0, P1, nN);
    k_init      <<<gNode, blk, 0, stream>>>(mBuf, denom, P2, nN);
    k_edge_logit<<<gEdge, blk, 0, stream>>>(P1, ei, beta1, mBuf, elog, nE, nN);
    k_edge_exp  <<<gEdge, blk, 0, stream>>>(ei, mBuf, denom, elog, nE, nN);
    k_edge_agg  <<<gEdge, blk, 0, stream>>>(ei, P0, denom, elog, P2, nE, nN);

    // ---- AGNN layer 2: P2 -> P0 ----
    k_norm      <<<gNode, blk, 0, stream>>>(P2, P1, nN);
    k_init      <<<gNode, blk, 0, stream>>>(mBuf, denom, P0, nN);
    k_edge_logit<<<gEdge, blk, 0, stream>>>(P1, ei, beta2, mBuf, elog, nE, nN);
    k_edge_exp  <<<gEdge, blk, 0, stream>>>(ei, mBuf, denom, elog, nE, nN);
    k_edge_agg  <<<gEdge, blk, 0, stream>>>(ei, P2, denom, elog, P0, nE, nN);

    // out = log_softmax(P0 @ W2 + b2)
    k_out<<<gOut, blk, 0, stream>>>(P0, W2, b2, (float*)d_out, nN);
}
